// GroupExpertChoiceMoELayer_55920474194570
// MI455X (gfx1250) — matrix operand order repaired and checked by reading, not compiled, not run on hardware
//
#include <hip/hip_runtime.h>
#include <hip/hip_bf16.h>
#include <stdint.h>

// Problem constants (from reference setup_inputs)
#define NT    8192      // B*S tokens
#define HDIM  2048      // hidden
#define NEXP  8         // experts
#define KCAP  1024      // capacity per expert
#define DFF   5461      // ffn dim
#define DFFP  5504      // padded to multiple of 64 (86 * 64)
#define LDA   40        // LDS row stride (bf16 elems): 32 data + 8 pad

typedef __attribute__((ext_vector_type(16))) __bf16 v16bf;
typedef __attribute__((ext_vector_type(8)))  float  v8f;

union FragAB { v16bf v; uint4 q[2]; };
union FragC  { v8f   v; float f[8]; };

#if __has_builtin(__builtin_amdgcn_global_load_async_to_lds_b128)
#define HAVE_ASYNC_LDS 1
#else
#define HAVE_ASYNC_LDS 0
#endif

// pointer types matching the builtin's signature: v4i in AS1 (global) / AS3 (LDS)
typedef int v4i_vec __attribute__((vector_size(16)));
typedef __attribute__((address_space(1))) v4i_vec* gp128_t;
typedef __attribute__((address_space(3))) v4i_vec* lp128_t;

// global -> LDS b128 copy: async DMA (ASYNCcnt) when available, else via VGPRs
__device__ __forceinline__ void copy_b128_g2l(const void* g, void* l) {
#if HAVE_ASYNC_LDS
  __builtin_amdgcn_global_load_async_to_lds_b128((gp128_t)g, (lp128_t)l, 0, 0);
#else
  *(uint4*)l = *(const uint4*)g;
#endif
}

__device__ __forceinline__ void wait_async_lds() {
#if HAVE_ASYNC_LDS
#if __has_builtin(__builtin_amdgcn_s_wait_asynccnt)
  __builtin_amdgcn_s_wait_asynccnt(0);
#else
  asm volatile("s_wait_asynccnt 0x0" ::: "memory");
#endif
#endif
}

__device__ __forceinline__ uint16_t f2bf(float f) {
  uint32_t x = __float_as_uint(f);
  x += 0x7FFFu + ((x >> 16) & 1u);   // round-to-nearest-even
  return (uint16_t)(x >> 16);
}

__device__ __forceinline__ float hash_uniform(uint32_t s) {
  s ^= 2747636419u; s *= 2654435769u;
  s ^= s >> 16;     s *= 2654435769u;
  s ^= s >> 16;     s *= 2654435769u;
  return (float)(s >> 8) * (1.0f / 16777216.0f) + 5.9604645e-8f;
}

// ---------------------------------------------------------------- zero out
__global__ void moe_zero(float* __restrict__ p, int n) {
  int i = blockIdx.x * blockDim.x + threadIdx.x;
  if (i < n) p[i] = 0.0f;
}

// ------------------------------------------------- router + gumbel + softmax
// One wave per token. Writes probs transposed: probsT[e*NT + t].
__global__ void __launch_bounds__(256) moe_router(
    const float* __restrict__ x, const float* __restrict__ rw,
    const float* __restrict__ rb, float* __restrict__ probsT)
{
  const int lane = threadIdx.x & 31;
  const int t = blockIdx.x * (blockDim.x >> 5) + (threadIdx.x >> 5);
  if (t >= NT) return;

  float acc[8];
#pragma unroll
  for (int e = 0; e < 8; ++e) acc[e] = 0.0f;

  const float* xr = x + (size_t)t * HDIM;
  for (int h = lane; h < HDIM; h += 32) {
    float xv = xr[h];
    const float4* w4 = (const float4*)&rw[h * 8];
    float4 w0 = w4[0], w1 = w4[1];
    acc[0] += xv * w0.x; acc[1] += xv * w0.y;
    acc[2] += xv * w0.z; acc[3] += xv * w0.w;
    acc[4] += xv * w1.x; acc[5] += xv * w1.y;
    acc[6] += xv * w1.z; acc[7] += xv * w1.w;
  }
#pragma unroll
  for (int e = 0; e < 8; ++e)
#pragma unroll
    for (int m = 16; m >= 1; m >>= 1)
      acc[e] += __shfl_xor(acc[e], m, 32);

  if (lane == 0) {
    float hl[8], mx = -3.0e38f;
#pragma unroll
    for (int e = 0; e < 8; ++e) {
      float u = hash_uniform((uint32_t)(t * 8 + e) * 2654435761u + 42u);
      float gum = -__logf(-__logf(u + 1e-10f) + 1e-10f);
      hl[e] = acc[e] + rb[e] + gum;              // ROUTER_TEMP == 1
      mx = fmaxf(mx, hl[e]);
    }
    float sum = 0.0f;
#pragma unroll
    for (int e = 0; e < 8; ++e) { hl[e] = __expf(hl[e] - mx); sum += hl[e]; }
    float inv = 1.0f / sum;
#pragma unroll
    for (int e = 0; e < 8; ++e) probsT[(size_t)e * NT + t] = hl[e] * inv;
  }
}

// --------------------------------------------- per-expert top-1024 of 8192
// One block of 1024 threads per expert; bitonic sort (descending) in LDS.
__global__ void __launch_bounds__(1024) moe_topk(
    const float* __restrict__ probsT, int* __restrict__ tk_idx,
    float* __restrict__ tk_w)
{
  __shared__ float    key[NT];     // 32 KB
  __shared__ uint16_t val[NT];     // 16 KB  (token idx < 8192 fits u16)
  const int e = blockIdx.x;
  const float* p = probsT + (size_t)e * NT;

  for (int i = threadIdx.x; i < NT; i += 1024) { key[i] = p[i]; val[i] = (uint16_t)i; }
  __syncthreads();

  for (int k2 = 2; k2 <= NT; k2 <<= 1) {
    for (int j = k2 >> 1; j >= 1; j >>= 1) {
      for (int i = threadIdx.x; i < NT; i += 1024) {
        int ixj = i ^ j;
        if (ixj > i) {
          bool desc = ((i & k2) == 0);
          float a = key[i], b = key[ixj];
          if (desc ? (a < b) : (a > b)) {
            key[i] = b; key[ixj] = a;
            uint16_t tv = val[i]; val[i] = val[ixj]; val[ixj] = tv;
          }
        }
      }
      __syncthreads();
    }
  }
  if (threadIdx.x < KCAP) {
    tk_idx[e * KCAP + threadIdx.x] = (int)val[threadIdx.x];
    tk_w  [e * KCAP + threadIdx.x] = key[threadIdx.x];
  }
}

// ---------------------- transpose + pad + f32->bf16 convert (weights -> ws)
// out[r*outCols + c] = bf16(in[c*inCols + r]) for c<inRows && r<inCols, else 0
__global__ void moe_transpose_bf16(
    const float* __restrict__ in, uint16_t* __restrict__ out,
    int inRows, int inCols, int outCols)
{
  __shared__ float tile[32][33];
  const int c0 = blockIdx.x * 32;   // out-col base (== in-row base)
  const int r0 = blockIdx.y * 32;   // out-row base (== in-col base)
#pragma unroll
  for (int i = threadIdx.y; i < 32; i += 8) {
    int c = c0 + i, r = r0 + threadIdx.x;
    float v = 0.0f;
    if (c < inRows && r < inCols) v = in[(size_t)c * inCols + r];
    tile[i][threadIdx.x] = v;
  }
  __syncthreads();
#pragma unroll
  for (int i = threadIdx.y; i < 32; i += 8) {
    int r = r0 + i, c = c0 + threadIdx.x;
    out[(size_t)r * outCols + c] = f2bf(tile[threadIdx.x][i]);
  }
}

// --------------------------------------------------------------- GEMM 1
// gate/up GEMM, fused SwiGLU. Block tile 128(M) x 64(N), Kstep 32,
// double-buffered LDS; weight tiles staged via async global->LDS DMA.
// 8 waves, each a 32x32 wave tile (2x2 WMMA frags) for BOTH gate and up.
__global__ void __launch_bounds__(256) moe_gemm1(
    const float* __restrict__ x, const int* __restrict__ tk_idx,
    const uint16_t* __restrict__ wg_t, const uint16_t* __restrict__ wu_t,
    const float* __restrict__ b_gate, const float* __restrict__ b_up,
    uint16_t* __restrict__ hbuf, int e)
{
  __shared__ uint16_t As[2 * 128 * LDA];   // gathered tokens, [m][k] bf16
  __shared__ uint16_t Bg[2 * 64 * LDA];    // gate weights,    [n][k] bf16
  __shared__ uint16_t Bu[2 * 64 * LDA];    // up weights,      [n][k] bf16
  __shared__ int tok[128];

  const int tid  = threadIdx.x;
  const int lane = tid & 31;
  const int wave = tid >> 5;
  const int wm = wave & 3, wn = wave >> 2;
  const int lg = lane >> 4, ln = lane & 15;
  const int mBase = blockIdx.y * 128;
  const int nBase = blockIdx.x * 64;
  const int brow = tid >> 2, bseg = tid & 3;

  if (tid < 128) tok[tid] = tk_idx[e * KCAP + mBase + tid];
  __syncthreads();

  FragC accG[2][2], accU[2][2];
#pragma unroll
  for (int mi = 0; mi < 2; ++mi)
#pragma unroll
    for (int ni = 0; ni < 2; ++ni)
#pragma unroll
      for (int v = 0; v < 8; ++v) { accG[mi][ni].f[v] = 0.f; accU[mi][ni].f[v] = 0.f; }

  auto stage = [&](int sk, int b) {
    // A: gather 128 token rows, convert f32->bf16 (4 float4 per thread)
#pragma unroll
    for (int t4 = 0; t4 < 4; ++t4) {
      int task = tid + t4 * 256;
      int row = task >> 3, seg = task & 7;
      const float4 vv = *(const float4*)&x[(size_t)tok[row] * HDIM + sk * 32 + seg * 4];
      uint2 dd;
      dd.x = (uint32_t)f2bf(vv.x) | ((uint32_t)f2bf(vv.y) << 16);
      dd.y = (uint32_t)f2bf(vv.z) | ((uint32_t)f2bf(vv.w) << 16);
      *(uint2*)&As[b * 128 * LDA + row * LDA + seg * 4] = dd;
    }
    // B: one async b128 per thread per matrix (pure bf16 copy, no VGPR round-trip)
    copy_b128_g2l(&wg_t[(size_t)(nBase + brow) * HDIM + sk * 32 + bseg * 8],
                  &Bg[b * 64 * LDA + brow * LDA + bseg * 8]);
    copy_b128_g2l(&wu_t[(size_t)(nBase + brow) * HDIM + sk * 32 + bseg * 8],
                  &Bu[b * 64 * LDA + brow * LDA + bseg * 8]);
  };

  const int steps = HDIM / 32;
  stage(0, 0);
  for (int kk = 0; kk < steps; ++kk) {
    const int cur = kk & 1;
    wait_async_lds();
    __syncthreads();                       // buffer `cur` ready for all waves
    if (kk + 1 < steps) stage(kk + 1, cur ^ 1);   // prefetch next under compute

    const uint16_t* Ab = &As[cur * 128 * LDA];
    const uint16_t* Gb = &Bg[cur * 64 * LDA];
    const uint16_t* Ub = &Bu[cur * 64 * LDA];

    FragAB a[2], bg[2], bu[2];
#pragma unroll
    for (int mi = 0; mi < 2; ++mi) {
      int r = wm * 32 + mi * 16 + ln;   // A: lane holds row r, K = {lg*8..}{16+lg*8..}
      a[mi].q[0] = *(const uint4*)&Ab[r * LDA + lg * 8];
      a[mi].q[1] = *(const uint4*)&Ab[r * LDA + lg * 8 + 16];
    }
#pragma unroll
    for (int ni = 0; ni < 2; ++ni) {
      int r = wn * 32 + ni * 16 + ln;   // B: lane holds col r, K = {lg*8..}{16+lg*8..}
      bg[ni].q[0] = *(const uint4*)&Gb[r * LDA + lg * 8];
      bg[ni].q[1] = *(const uint4*)&Gb[r * LDA + lg * 8 + 16];
      bu[ni].q[0] = *(const uint4*)&Ub[r * LDA + lg * 8];
      bu[ni].q[1] = *(const uint4*)&Ub[r * LDA + lg * 8 + 16];
    }
#pragma unroll
    for (int mi = 0; mi < 2; ++mi)
#pragma unroll
      for (int ni = 0; ni < 2; ++ni) {
        accG[mi][ni].v = __builtin_amdgcn_wmma_f32_16x16x32_bf16(
            false, a[mi].v, false, bg[ni].v, (short)0, accG[mi][ni].v, false, false);
        accU[mi][ni].v = __builtin_amdgcn_wmma_f32_16x16x32_bf16(
            false, a[mi].v, false, bu[ni].v, (short)0, accU[mi][ni].v, false, false);
      }
    __syncthreads();                       // readers done before buffer reuse
  }

  // Fused SwiGLU epilogue -> bf16 h
#pragma unroll
  for (int mi = 0; mi < 2; ++mi) {
#pragma unroll
    for (int ni = 0; ni < 2; ++ni) {
      int gn = nBase + wn * 32 + ni * 16 + ln;
      float bgv = (gn < DFF) ? b_gate[gn] : 0.f;
      float buv = (gn < DFF) ? b_up[gn]   : 0.f;
#pragma unroll
      for (int v = 0; v < 8; ++v) {
        int gm = mBase + wm * 32 + mi * 16 + lg * 8 + v;
        float g = accG[mi][ni].f[v] + bgv;
        float u = accU[mi][ni].f[v] + buv;
        float s = g / (1.0f + __expf(-g));
        hbuf[(size_t)gm * DFFP + gn] = f2bf(s * u);
      }
    }
  }
}

// --------------------------------------------------------------- GEMM 2
// down-proj + weight by G + scatter-add (atomic f32) into out.
// Both A (h, bf16) and B (wd_t, bf16) tiles staged via async global->LDS DMA.
__global__ void __launch_bounds__(256) moe_gemm2(
    const uint16_t* __restrict__ hbuf, const uint16_t* __restrict__ wd_t,
    const float* __restrict__ b_down, const int* __restrict__ tk_idx,
    const float* __restrict__ tk_w, float* __restrict__ out, int e)
{
  __shared__ uint16_t As[2 * 128 * LDA];
  __shared__ uint16_t Bd[2 * 64 * LDA];

  const int tid  = threadIdx.x;
  const int lane = tid & 31;
  const int wave = tid >> 5;
  const int wm = wave & 3, wn = wave >> 2;
  const int lg = lane >> 4, ln = lane & 15;
  const int mBase = blockIdx.y * 128;
  const int nBase = blockIdx.x * 64;
  const int brow = tid >> 2, bseg = tid & 3;

  FragC acc[2][2];
#pragma unroll
  for (int mi = 0; mi < 2; ++mi)
#pragma unroll
    for (int ni = 0; ni < 2; ++ni)
#pragma unroll
      for (int v = 0; v < 8; ++v) acc[mi][ni].f[v] = 0.f;

  auto stage = [&](int sk, int b) {
#pragma unroll
    for (int t4 = 0; t4 < 2; ++t4) {
      int task = tid + t4 * 256;
      int row = task >> 2, seg = task & 3;
      copy_b128_g2l(&hbuf[(size_t)(mBase + row) * DFFP + sk * 32 + seg * 8],
                    &As[b * 128 * LDA + row * LDA + seg * 8]);
    }
    copy_b128_g2l(&wd_t[(size_t)(nBase + brow) * DFFP + sk * 32 + bseg * 8],
                  &Bd[b * 64 * LDA + brow * LDA + bseg * 8]);
  };

  const int steps = DFFP / 32;
  stage(0, 0);
  for (int kk = 0; kk < steps; ++kk) {
    const int cur = kk & 1;
    wait_async_lds();
    __syncthreads();
    if (kk + 1 < steps) stage(kk + 1, cur ^ 1);

    const uint16_t* Ab = &As[cur * 128 * LDA];
    const uint16_t* Db = &Bd[cur * 64 * LDA];

    FragAB a[2], b[2];
#pragma unroll
    for (int mi = 0; mi < 2; ++mi) {
      int r = wm * 32 + mi * 16 + ln;
      a[mi].q[0] = *(const uint4*)&Ab[r * LDA + lg * 8];
      a[mi].q[1] = *(const uint4*)&Ab[r * LDA + lg * 8 + 16];
    }
#pragma unroll
    for (int ni = 0; ni < 2; ++ni) {
      int r = wn * 32 + ni * 16 + ln;
      b[ni].q[0] = *(const uint4*)&Db[r * LDA + lg * 8];
      b[ni].q[1] = *(const uint4*)&Db[r * LDA + lg * 8 + 16];
    }
#pragma unroll
    for (int mi = 0; mi < 2; ++mi)
#pragma unroll
      for (int ni = 0; ni < 2; ++ni)
        acc[mi][ni].v = __builtin_amdgcn_wmma_f32_16x16x32_bf16(
            false, a[mi].v, false, b[ni].v, (short)0, acc[mi][ni].v, false, false);
    __syncthreads();
  }

  // Epilogue: (acc + b_down) * G -> atomic scatter-add to token rows
#pragma unroll
  for (int mi = 0; mi < 2; ++mi) {
    int tdst[8]; float gw[8];
#pragma unroll
    for (int v = 0; v < 8; ++v) {
      int gm = mBase + wm * 32 + mi * 16 + lg * 8 + v;
      tdst[v] = tk_idx[e * KCAP + gm];
      gw[v]   = tk_w [e * KCAP + gm];
    }
#pragma unroll
    for (int ni = 0; ni < 2; ++ni) {
      int gn = nBase + wn * 32 + ni * 16 + ln;
      float bd = b_down[gn];
#pragma unroll
      for (int v = 0; v < 8; ++v) {
        float val = (acc[mi][ni].f[v] + bd) * gw[v];
        atomicAdd(&out[(size_t)tdst[v] * HDIM + gn], val);
      }
    }
  }
}

// ------------------------------------------------------------------ launch
extern "C" void kernel_launch(void* const* d_in, const int* in_sizes, int n_in,
                              void* d_out, int out_size, void* d_ws, size_t ws_size,
                              hipStream_t stream) {
  (void)in_sizes; (void)n_in; (void)out_size; (void)ws_size;
  const float* x        = (const float*)d_in[0];
  const float* router_w = (const float*)d_in[1];
  const float* router_b = (const float*)d_in[2];
  const float* w_gate   = (const float*)d_in[3];
  const float* b_gate   = (const float*)d_in[4];
  const float* w_up     = (const float*)d_in[5];
  const float* b_up     = (const float*)d_in[6];
  const float* w_down   = (const float*)d_in[7];
  const float* b_down   = (const float*)d_in[8];
  float* out = (float*)d_out;

  char* ws = (char*)d_ws;
  float*    probsT = (float*)(ws);                         // 8*8192*4      = 262144
  int*      tk_idx = (int*)  (ws + 262144);                // 8*1024*4      = 32768
  float*    tk_w   = (float*)(ws + 294912);                // 8*1024*4      = 32768
  uint16_t* wg_t   = (uint16_t*)(ws + 327680);             // 5504*2048*2   = 22544384
  uint16_t* wu_t   = (uint16_t*)(ws + 22872064);           // 22544384
  uint16_t* wd_t   = (uint16_t*)(ws + 45416448);           // 2048*5504*2   = 22544384
  uint16_t* hbuf   = (uint16_t*)(ws + 67960832);           // 1024*5504*2   = 11272192

  // 1) zero output (harness poisons it)
  moe_zero<<<dim3((NT * HDIM + 255) / 256), dim3(256), 0, stream>>>(out, NT * HDIM);

  // 2) router + gumbel + softmax (transposed probs)
  moe_router<<<dim3(NT / 8), dim3(256), 0, stream>>>(x, router_w, router_b, probsT);

  // 3) per-expert top-1024
  moe_topk<<<dim3(NEXP), dim3(1024), 0, stream>>>(probsT, tk_idx, tk_w);

  // 4) weight transpose + pad + bf16 convert (N-major, K-contiguous rows)
  moe_transpose_bf16<<<dim3(HDIM / 32, DFFP / 32), dim3(32, 8), 0, stream>>>(
      w_gate, wg_t, HDIM, DFF, HDIM);
  moe_transpose_bf16<<<dim3(HDIM / 32, DFFP / 32), dim3(32, 8), 0, stream>>>(
      w_up, wu_t, HDIM, DFF, HDIM);
  moe_transpose_bf16<<<dim3(DFFP / 32, HDIM / 32), dim3(32, 8), 0, stream>>>(
      w_down, wd_t, DFF, HDIM, DFFP);

  // 5) per-expert fused SwiGLU FFN + weighted scatter-add
  for (int e = 0; e < NEXP; ++e) {
    moe_gemm1<<<dim3(DFFP / 64, KCAP / 128), dim3(256), 0, stream>>>(
        x, tk_idx, wg_t, wu_t, b_gate, b_up, hbuf, e);
    moe_gemm2<<<dim3(HDIM / 64, KCAP / 128), dim3(256), 0, stream>>>(
        hbuf, wd_t, b_down, tk_idx, tk_w, out, e);
  }
}
